// Bert_31988916420705
// MI455X (gfx1250) — matrix operand order repaired and checked by reading, not compiled, not run on hardware
//
#include <hip/hip_runtime.h>

// ---------------- model constants ----------------
#define VOCAB 30522
#define DMODEL 768
#define NHEAD 12
#define NLAYER 12
#define DFF 3072
#define SEQ 512
#define BATCH 4
#define HEADDIM 64
#define LNEPS 1e-5f

// ---------------- WMMA vector types ----------------
typedef __attribute__((ext_vector_type(16))) __bf16 v16bf;
typedef __attribute__((ext_vector_type(8)))  __bf16 v8bf;
typedef __attribute__((ext_vector_type(8)))  float  v8f;

// ---- fragment loads from LDS (ISA 7.12.2 layouts, wave32) ----
// A tile in LDS: [128 rows][32 k] bf16. 16-bit A 16x32: lane<16 -> K 0..7,16..23 ; lane>=16 -> K 8..15,24..31
__device__ __forceinline__ v16bf load_a_frag(const __bf16* sA, int mbase) {
    int lane = threadIdx.x & 31;
    int m = mbase + (lane & 15);
    int kb = (lane >> 4) * 8;
    const v8bf* p = (const v8bf*)(sA + m * 32 + kb);
    union { v16bf v; v8bf h[2]; } u;
    u.h[0] = p[0];      // K = kb .. kb+7
    u.h[1] = p[2];      // K = kb+16 .. kb+23
    return u.v;
}
// B tile in LDS transposed: [128 n][32 k] bf16. 16-bit B 32x16: lane<16 -> K 0..7,16..23 ; lane>=16 -> K 8..15,24..31, N=lane&15
__device__ __forceinline__ v16bf load_b_frag(const __bf16* sBt, int nbase) {
    int lane = threadIdx.x & 31;
    int n = nbase + (lane & 15);
    int kb = (lane >> 4) * 8;
    const v8bf* p = (const v8bf*)(sBt + n * 32 + kb);
    union { v16bf v; v8bf h[2]; } u;
    u.h[0] = p[0];
    u.h[1] = p[2];
    return u.v;
}

__device__ __forceinline__ float gelu_exact(float x) {
    return 0.5f * x * (1.0f + erff(x * 0.70710678118654752f));
}

// ---------------- batched GEMM: C = act(alpha * A*B + bias) ----------------
// Block tile 128x128, 8 waves, wave tile 32x64 (2x4 WMMA tiles), K-step 32.
// Assumptions honored by every launch below: M % 128 == 0, K % 32 == 0, lda % 4 == 0,
// (transB=1 => ldb % 4 == 0). N may be ragged (logits) -> guarded in B staging + epilogue.
__global__ void __launch_bounds__(256)
gemm_bf16_wmma(const float* __restrict__ A, const float* __restrict__ Bm,
               float* __restrict__ C,
               int M, int N, int K, int lda, int ldb, int ldc, int transB,
               long long sAo, long long sAi, long long sBo, long long sBi,
               long long sCo, long long sCi, int inner,
               const float* __restrict__ bias, int act, float alpha)
{
    __shared__ __align__(16) __bf16 sA[128 * 32];
    __shared__ __align__(16) __bf16 sB[128 * 32];   // stored transposed: [n][k]

    int z  = blockIdx.z;
    int zo = z / inner, zi = z % inner;
    A  += zo * sAo + zi * sAi;
    Bm += zo * sBo + zi * sBi;
    C  += zo * sCo + zi * sCi;

    const int m0  = blockIdx.y * 128;
    const int n0  = blockIdx.x * 128;
    const int tid = threadIdx.x;
    const int wave = tid >> 5;
    const int lane = tid & 31;
    const int wm = (wave >> 1) * 32;   // wave tile origin inside block tile (4 rows)
    const int wn = (wave & 1) * 64;    // (2 cols of 64)

    v8f acc[2][4];
    {
        const v8f vz = {0.f,0.f,0.f,0.f,0.f,0.f,0.f,0.f};
        #pragma unroll
        for (int a = 0; a < 2; ++a)
            #pragma unroll
            for (int b = 0; b < 4; ++b) acc[a][b] = vz;
    }

    for (int k0 = 0; k0 < K; k0 += 32) {
        __syncthreads();
        // ---- stage A 128x32 (4096 elems, fp32 -> bf16, fully in-range) ----
        #pragma unroll
        for (int i = 0; i < 4; ++i) {
            int f = i * 1024 + tid * 4;
            int r = f >> 5, c = f & 31;
            const float* src = A + (long long)(m0 + r) * lda + (k0 + c);
            float4 t4 = *(const float4*)src;
            if (i == 0 && k0 + 32 < K) __builtin_prefetch(src + 32, 0, 3);  // global_prefetch_b8
            __bf16* d = &sA[r * 32 + c];
            d[0] = (__bf16)t4.x; d[1] = (__bf16)t4.y; d[2] = (__bf16)t4.z; d[3] = (__bf16)t4.w;
        }
        // ---- stage B into transposed [n][k] tile ----
        if (!transB) {
            // B is KxN row-major; read 4 consecutive n (coalesced), scatter to 4 LDS rows
            #pragma unroll
            for (int i = 0; i < 4; ++i) {
                int f = i * 1024 + tid * 4;
                int kk = f >> 7, nn = f & 127;
                int gn = n0 + nn;
                const float* src = Bm + (long long)(k0 + kk) * ldb + gn;
                if (((ldb & 3) == 0) && (gn + 3 < N)) {
                    float4 t4 = *(const float4*)src;
                    sB[(nn + 0) * 32 + kk] = (__bf16)t4.x;
                    sB[(nn + 1) * 32 + kk] = (__bf16)t4.y;
                    sB[(nn + 2) * 32 + kk] = (__bf16)t4.z;
                    sB[(nn + 3) * 32 + kk] = (__bf16)t4.w;
                } else {
                    #pragma unroll
                    for (int j = 0; j < 4; ++j)
                        sB[(nn + j) * 32 + kk] = (gn + j < N) ? (__bf16)src[j] : (__bf16)0.f;
                }
            }
        } else {
            // B is NxK row-major (used for Q*K^T); contiguous along k
            #pragma unroll
            for (int i = 0; i < 4; ++i) {
                int f = i * 1024 + tid * 4;
                int r = f >> 5, c = f & 31;
                int gn = n0 + r;
                float4 t4 = {0.f, 0.f, 0.f, 0.f};
                if (gn < N) t4 = *(const float4*)(Bm + (long long)gn * ldb + (k0 + c));
                __bf16* d = &sB[r * 32 + c];
                d[0] = (__bf16)t4.x; d[1] = (__bf16)t4.y; d[2] = (__bf16)t4.z; d[3] = (__bf16)t4.w;
            }
        }
        __syncthreads();

        // ---- 32x64 wave tile = 2x4 WMMA tiles ----
        v16bf af[2], bfr[4];
        af[0] = load_a_frag(sA, wm + 0);
        af[1] = load_a_frag(sA, wm + 16);
        #pragma unroll
        for (int fn = 0; fn < 4; ++fn) bfr[fn] = load_b_frag(sB, wn + fn * 16);
        #pragma unroll
        for (int fm = 0; fm < 2; ++fm)
            #pragma unroll
            for (int fn = 0; fn < 4; ++fn)
                acc[fm][fn] = __builtin_amdgcn_wmma_f32_16x16x32_bf16(
                    false, af[fm], false, bfr[fn], (short)0, acc[fm][fn], false, false);
    }

    // ---- epilogue: alpha, bias, act, store fp32 (rows always in range; cols guarded) ----
    #pragma unroll
    for (int fm = 0; fm < 2; ++fm) {
        #pragma unroll
        for (int fn = 0; fn < 4; ++fn) {
            int col = n0 + wn + fn * 16 + (lane & 15);
            if (col < N) {
                float bv = bias ? bias[col] : 0.f;
                #pragma unroll
                for (int r = 0; r < 8; ++r) {
                    int row = m0 + wm + fm * 16 + r + ((lane >> 4) << 3);
                    float v = acc[fm][fn][r] * alpha + bv;
                    if (act) v = gelu_exact(v);
                    C[(long long)row * ldc + col] = v;
                }
            }
        }
    }
}

// ---------------- embeddings: word + pos + seg ----------------
__global__ void __launch_bounds__(256)
embed_kernel(const int* __restrict__ ids, const float* __restrict__ we,
             const float* __restrict__ pe, const float* __restrict__ se,
             float* __restrict__ out)
{
    long long r = blockIdx.x;            // b*SEQ + s
    int sidx = (int)(r % SEQ);
    long long id = ids[r];
    int t = threadIdx.x;
    #pragma unroll
    for (int i = 0; i < 3; ++i) {
        int j = t + i * 256;
        out[r * DMODEL + j] = we[id * DMODEL + j] + pe[(long long)sidx * DMODEL + j] + se[j];
    }
}

// ---------------- LayerNorm( x (+res) ) ----------------
__global__ void __launch_bounds__(256)
layernorm_kernel(const float* __restrict__ x, const float* __restrict__ res,
                 const float* __restrict__ g, const float* __restrict__ b,
                 float* __restrict__ out)
{
    __shared__ float red[256];
    long long r = blockIdx.x;
    const float* xr = x + r * DMODEL;
    const float* rr = res ? res + r * DMODEL : nullptr;
    int t = threadIdx.x;
    float v[3]; float s = 0.f;
    #pragma unroll
    for (int i = 0; i < 3; ++i) {
        int j = t + i * 256;
        v[i] = xr[j] + (rr ? rr[j] : 0.f);
        s += v[i];
    }
    red[t] = s; __syncthreads();
    for (int k = 128; k > 0; k >>= 1) { if (t < k) red[t] += red[t + k]; __syncthreads(); }
    float mu = red[0] * (1.0f / DMODEL); __syncthreads();
    float sq = 0.f;
    #pragma unroll
    for (int i = 0; i < 3; ++i) { float d = v[i] - mu; sq += d * d; }
    red[t] = sq; __syncthreads();
    for (int k = 128; k > 0; k >>= 1) { if (t < k) red[t] += red[t + k]; __syncthreads(); }
    float rs = rsqrtf(red[0] * (1.0f / DMODEL) + LNEPS);
    #pragma unroll
    for (int i = 0; i < 3; ++i) {
        int j = t + i * 256;
        out[r * DMODEL + j] = (v[i] - mu) * rs * g[j] + b[j];
    }
}

// ---------------- masked softmax over rows of length SEQ, in place ----------------
__global__ void __launch_bounds__(256)
softmax_kernel(float* __restrict__ p, const int* __restrict__ amask, int rowsPerB)
{
    __shared__ float red[256];
    long long rid = blockIdx.x;
    float* row = p + rid * (long long)SEQ;
    int b = (int)(rid / rowsPerB);
    const int* mrow = amask + (long long)b * SEQ;
    int t = threadIdx.x;
    float vals[2]; float mx = -INFINITY;
    #pragma unroll
    for (int i = 0; i < 2; ++i) {
        int j = t + i * 256;
        float v = row[j];
        if (mrow[j] == 0) v = -INFINITY;
        vals[i] = v;
        mx = fmaxf(mx, v);
    }
    red[t] = mx; __syncthreads();
    for (int k = 128; k > 0; k >>= 1) { if (t < k) red[t] = fmaxf(red[t], red[t + k]); __syncthreads(); }
    mx = red[0]; __syncthreads();
    float s = 0.f;
    #pragma unroll
    for (int i = 0; i < 2; ++i) { vals[i] = expf(vals[i] - mx); s += vals[i]; }
    red[t] = s; __syncthreads();
    for (int k = 128; k > 0; k >>= 1) { if (t < k) red[t] += red[t + k]; __syncthreads(); }
    float inv = 1.0f / red[0];
    #pragma unroll
    for (int i = 0; i < 2; ++i) row[t + i * 256] = vals[i] * inv;
}

// ---------------- host-side orchestration ----------------
extern "C" void kernel_launch(void* const* d_in, const int* in_sizes, int n_in,
                              void* d_out, int out_size, void* d_ws, size_t ws_size,
                              hipStream_t stream)
{
    (void)in_sizes; (void)n_in; (void)out_size; (void)ws_size;

    const int*   ids   = (const int*)  d_in[0];
    const int*   amask = (const int*)  d_in[1];
    const float* we    = (const float*)d_in[2];
    const float* pe    = (const float*)d_in[3];
    const float* se    = (const float*)d_in[4];
    const float* elg   = (const float*)d_in[5];
    const float* elb   = (const float*)d_in[6];
    const float* Wq    = (const float*)d_in[7];
    const float* Wk    = (const float*)d_in[8];
    const float* Wv    = (const float*)d_in[9];
    const float* Wo    = (const float*)d_in[10];
    const float* f1w   = (const float*)d_in[11];
    const float* f1b   = (const float*)d_in[12];
    const float* f2w   = (const float*)d_in[13];
    const float* f2b   = (const float*)d_in[14];
    const float* g1    = (const float*)d_in[15];
    const float* b1    = (const float*)d_in[16];
    const float* g2    = (const float*)d_in[17];
    const float* b2    = (const float*)d_in[18];
    const float* outw  = (const float*)d_in[19];
    const float* outb  = (const float*)d_in[20];

    float* out = (float*)d_out;

    const int Mtok = BATCH * SEQ;                         // 2048
    const size_t logitsElems = (size_t)Mtok * VOCAB;
    const size_t attnPerL    = (size_t)BATCH * NHEAD * SEQ * SEQ;

    // workspace carve-up (fp32)
    float* ws = (float*)d_ws;
    size_t off = 0;
    auto alloc = [&](size_t n) { float* p = ws + off; off += n; return p; };
    float* bx   = alloc((size_t)Mtok * DMODEL);
    float* bq   = alloc((size_t)Mtok * DMODEL);
    float* bk   = alloc((size_t)Mtok * DMODEL);
    float* bv   = alloc((size_t)Mtok * DMODEL);
    float* bctx = alloc((size_t)Mtok * DMODEL);
    float* bo   = alloc((size_t)Mtok * DMODEL);
    float* bx1  = alloc((size_t)Mtok * DMODEL);
    float* bf2  = alloc((size_t)Mtok * DMODEL);
    float* bh   = alloc((size_t)Mtok * DFF);

    auto gemm = [&](const float* A, const float* Bm, float* C,
                    int M, int N, int K, int lda, int ldb, int ldc, int tB,
                    long long sAo, long long sAi, long long sBo, long long sBi,
                    long long sCo, long long sCi, int inner, int nb,
                    const float* bias, int act, float alpha) {
        dim3 grid((N + 127) / 128, (M + 127) / 128, nb);
        gemm_bf16_wmma<<<grid, 256, 0, stream>>>(A, Bm, C, M, N, K, lda, ldb, ldc, tB,
                                                 sAo, sAi, sBo, sBi, sCo, sCi, inner,
                                                 bias, act, alpha);
    };

    // embeddings + LN  (bq used as temp before layer 0 overwrites it)
    embed_kernel<<<Mtok, 256, 0, stream>>>(ids, we, pe, se, bq);
    layernorm_kernel<<<Mtok, 256, 0, stream>>>(bq, nullptr, elg, elb, bx);

    const float scale = 1.0f / sqrtf((float)DMODEL);   // embed dim, matches reference

    for (int l = 0; l < NLAYER; ++l) {
        const float* wq = Wq + (size_t)l * DMODEL * DMODEL;
        const float* wk = Wk + (size_t)l * DMODEL * DMODEL;
        const float* wv = Wv + (size_t)l * DMODEL * DMODEL;
        const float* wo = Wo + (size_t)l * DMODEL * DMODEL;
        const float* w1 = f1w + (size_t)l * DMODEL * DFF;
        const float* bb1 = f1b + (size_t)l * DFF;
        const float* w2 = f2w + (size_t)l * DFF * DMODEL;
        const float* bb2 = f2b + (size_t)l * DMODEL;
        float* attnL = out + logitsElems + (size_t)l * attnPerL;

        // Q,K,V projections (no bias in reference)
        gemm(bx, wq, bq, Mtok, DMODEL, DMODEL, DMODEL, DMODEL, DMODEL, 0,
             0,0, 0,0, 0,0, 1, 1, nullptr, 0, 1.0f);
        gemm(bx, wk, bk, Mtok, DMODEL, DMODEL, DMODEL, DMODEL, DMODEL, 0,
             0,0, 0,0, 0,0, 1, 1, nullptr, 0, 1.0f);
        gemm(bx, wv, bv, Mtok, DMODEL, DMODEL, DMODEL, DMODEL, DMODEL, 0,
             0,0, 0,0, 0,0, 1, 1, nullptr, 0, 1.0f);

        // scores[b,h] = scale * Q_bh (S x HD) * K_bh^T (HD x S) -> attnL, batched over b*h
        gemm(bq, bk, attnL, SEQ, SEQ, HEADDIM, DMODEL, DMODEL, SEQ, 1,
             (long long)SEQ * DMODEL, HEADDIM,
             (long long)SEQ * DMODEL, HEADDIM,
             (long long)NHEAD * SEQ * SEQ, (long long)SEQ * SEQ,
             NHEAD, BATCH * NHEAD, nullptr, 0, scale);

        // softmax in place (this region is also a model output)
        softmax_kernel<<<BATCH * NHEAD * SEQ, 256, 0, stream>>>(attnL, amask, NHEAD * SEQ);

        // ctx[b,h] = attn (S x S) * V_bh (S x HD)
        gemm(attnL, bv, bctx, SEQ, HEADDIM, SEQ, SEQ, DMODEL, DMODEL, 0,
             (long long)NHEAD * SEQ * SEQ, (long long)SEQ * SEQ,
             (long long)SEQ * DMODEL, HEADDIM,
             (long long)SEQ * DMODEL, HEADDIM,
             NHEAD, BATCH * NHEAD, nullptr, 0, 1.0f);

        // o = ctx @ Wo
        gemm(bctx, wo, bo, Mtok, DMODEL, DMODEL, DMODEL, DMODEL, DMODEL, 0,
             0,0, 0,0, 0,0, 1, 1, nullptr, 0, 1.0f);

        // x1 = LN(x + o)
        layernorm_kernel<<<Mtok, 256, 0, stream>>>(bx, bo, g1 + (size_t)l * DMODEL,
                                                   b1 + (size_t)l * DMODEL, bx1);

        // h = gelu(x1 @ W1 + b1)
        gemm(bx1, w1, bh, Mtok, DFF, DMODEL, DMODEL, DFF, DFF, 0,
             0,0, 0,0, 0,0, 1, 1, bb1, 1, 1.0f);

        // f2 = h @ W2 + b2
        gemm(bh, w2, bf2, Mtok, DMODEL, DFF, DFF, DMODEL, DMODEL, 0,
             0,0, 0,0, 0,0, 1, 1, bb2, 0, 1.0f);

        // x = LN(x1 + f2)
        layernorm_kernel<<<Mtok, 256, 0, stream>>>(bx1, bf2, g2 + (size_t)l * DMODEL,
                                                   b2 + (size_t)l * DMODEL, bx);
    }

    // logits = x @ out_w + out_b   (N = 30522, ragged edge handled in kernel)
    gemm(bx, outw, out, Mtok, VOCAB, DMODEL, DMODEL, VOCAB, VOCAB, 0,
         0,0, 0,0, 0,0, 1, 1, outb, 0, 1.0f);
}
